// STNRNN_84301618086388
// MI455X (gfx1250) — matrix operand order repaired and checked by reading, not compiled, not run on hardware
//
#include <hip/hip_runtime.h>
#include <hip/hip_bf16.h>
#include <cstdint>
#include <cstddef>

// Problem constants (match reference)
#define BATCH   32
#define T_STEPS 2048
#define INP     64
#define HID     512

#define BDIM    256     // 8 waves of 32
#define NWG     16      // persistent workgroups; each owns 128 gate columns (32 h cols x 4 gates)

// ---- dynamic LDS layout (bytes) ----
#define OFF_W   0                       // 147456 B : 144 B-fragment tiles * 256 dwords
#define OFF_H   147456                  // h_t staged bf16: 32 rows x stride 520 -> 33280 B
#define OFF_X   180736                  // x_t staged bf16: 32 rows x stride 72  -> 4608 B
#define OFF_G   185344                  // gates f32: 32 x stride 132           -> 16896 B
#define OFF_SH  202240                  // relaxed h state f32 (32x32)          -> 4096 B
#define OFF_SC  206336                  // relaxed c state f32 (32x32)          -> 4096 B
#define OFF_B   210432                  // combined bias f32 (128)              -> 512 B
#define SMEM_BYTES 210944

typedef __attribute__((ext_vector_type(16))) __bf16        v16bf;
typedef __attribute__((ext_vector_type(8)))  float         v8f;
typedef __attribute__((ext_vector_type(8)))  unsigned int  v8u;
typedef __attribute__((ext_vector_type(4)))  unsigned int  u32x4;
typedef __attribute__((ext_vector_type(4)))  int           i32x4;
typedef __attribute__((ext_vector_type(8)))  int           i32x8;

__device__ __forceinline__ unsigned short f2bf(float f) {
  unsigned u = __builtin_bit_cast(unsigned, f);
  u += 0x7FFFu + ((u >> 16) & 1u);           // round-to-nearest-even
  return (unsigned short)(u >> 16);
}
__device__ __forceinline__ unsigned packbf(float a, float b) {
  return (unsigned)f2bf(a) | ((unsigned)f2bf(b) << 16);
}
__device__ __forceinline__ float sigm(float x) { return 1.0f / (1.0f + __expf(-x)); }

// Grid-wide barrier: monotonically increasing counter in d_ws.
__device__ __forceinline__ void grid_sync(unsigned* cnt, unsigned target) {
  __threadfence();
  __syncthreads();
  if (threadIdx.x == 0) {
    __hip_atomic_fetch_add(cnt, 1u, __ATOMIC_RELEASE, __HIP_MEMORY_SCOPE_AGENT);
    while (__hip_atomic_load(cnt, __ATOMIC_ACQUIRE, __HIP_MEMORY_SCOPE_AGENT) < target) {
      __builtin_amdgcn_s_sleep(2);
    }
  }
  __syncthreads();
  __threadfence();
}

__global__ void stn_init(unsigned char* ws) {
  unsigned* cnt = (unsigned*)ws;
  unsigned short* hb = (unsigned short*)(ws + 256);
  int idx = blockIdx.x * blockDim.x + threadIdx.x;
  if (idx == 0) *cnt = 0u;
  for (int i = idx; i < 2 * BATCH * HID; i += gridDim.x * blockDim.x) hb[i] = 0;
}

__global__ __launch_bounds__(BDIM, 1)
void stn_lstm_persistent(const float* __restrict__ x,
                         const float* __restrict__ W_ih,
                         const float* __restrict__ W_hh,
                         const float* __restrict__ b_ih,
                         const float* __restrict__ b_hh,
                         float* __restrict__ out,
                         unsigned char* __restrict__ ws)
{
  extern __shared__ unsigned char smem[];
  unsigned*       wlds = (unsigned*)(smem + OFF_W);
  unsigned short* hb   = (unsigned short*)(smem + OFF_H);
  unsigned short* xb   = (unsigned short*)(smem + OFF_X);
  float*          gl   = (float*)(smem + OFF_G);
  float*          sh   = (float*)(smem + OFF_SH);
  float*          sc   = (float*)(smem + OFF_SC);
  float*          bl   = (float*)(smem + OFF_B);

  unsigned*       cnt   = (unsigned*)ws;
  unsigned short* hbuf0 = (unsigned short*)(ws + 256);
  unsigned short* hbuf1 = (unsigned short*)(ws + 256 + BATCH * HID * 2);

  const int wg   = blockIdx.x;     // owns gate cols { q*512 + wg*32 + [0,32) : q=0..3 }
  const int tid  = threadIdx.x;
  const int lane = tid & 31;
  const int wv   = tid >> 5;       // wave id 0..7; wave covers 16 gate cols

  // ---------- one-time: pack weight slice (bf16) into LDS in WMMA B-fragment order ----------
  // B fragment (32x16, K-major): dword v of lane L holds K = {2v, 2v+1} + (L>=16 ? 8 : 0) + (v>=4 ? 8 : 0), col = L&15
  for (int i = 0; i < 144; ++i) {
    int E    = tid + i * BDIM;            // 0..36863
    int v    = E & 7;
    int ln   = (E >> 3) & 31;
    int tk   = E >> 8;                    // tile id = w*18 + kc, 0..143
    int kc   = tk % 18;
    int w    = tk / 18;
    int gcol = w * 16 + (ln & 15);        // local gate column 0..127
    int grow = (gcol >> 5) * HID + wg * 32 + (gcol & 31);  // global weight row
    int kb   = 2 * v + ((ln >> 4) << 3) + ((v >> 2) << 3);
    int k    = kc * 32 + kb;              // 0..575: [0,512)=h via W_hh, [512,576)=x via W_ih
    float f0, f1;
    if (k < HID) {
      const float* p = W_hh + (size_t)grow * HID + k;
      f0 = p[0]; f1 = p[1];
    } else {
      const float* p = W_ih + (size_t)grow * INP + (k - HID);
      f0 = p[0]; f1 = p[1];
    }
    wlds[(size_t)tk * 256 + ln * 8 + v] = packbf(f0, f1);
  }
  if (tid < 128) {
    int grow = (tid >> 5) * HID + wg * 32 + (tid & 31);
    bl[tid] = b_ih[grow] + b_hh[grow];
  }
  for (int i = tid; i < BATCH * 32; i += BDIM) { sh[i] = 0.f; sc[i] = 0.f; }
  __syncthreads();

  const int row0  = lane & 15;
  const int khalf = lane >> 4;
  const int mtop  = khalf * 8;
  const int ncol  = wv * 16 + row0;      // this lane's C column (local gate col)
  const unsigned* wbase = wlds + (size_t)(wv * 18) * 256 + lane * 8;
  const unsigned short* ap0 = hb + row0 * 520 + khalf * 8;          // A rows 0..15
  const unsigned short* ap1 = hb + (16 + row0) * 520 + khalf * 8;   // A rows 16..31
  const unsigned short* axp0 = xb + row0 * 72 + khalf * 8;
  const unsigned short* axp1 = xb + (16 + row0) * 72 + khalf * 8;
  const size_t finalOff = (size_t)BATCH * T_STEPS * HID;
  const unsigned ldsH = __builtin_amdgcn_groupstaticsize() + OFF_H; // TDM dest LDS byte offset

  for (int t = 0; t < T_STEPS; ++t) {
    const unsigned short* rbuf = (t & 1) ? hbuf1 : hbuf0;
    unsigned short*       wbuf = (t & 1) ? hbuf0 : hbuf1;

    // ---- stage h_t (bf16, 32x512) into padded LDS via Tensor Data Mover ----
    // 2D tile: tile_dim0 = 512 elems of 2B per row, 32 rows; LDS padding of 4 dwords
    // after every 256 dwords reproduces the 520-bf16 padded row stride.
    if (wv == 0) {
      unsigned long long ga = (unsigned long long)(uintptr_t)rbuf;
      u32x4 g0;
      g0[0] = 1u;                                            // count=1 (user descriptor)
      g0[1] = ldsH;                                          // lds_addr
      g0[2] = (unsigned)(ga & 0xFFFFFFFFu);                  // global_addr[31:0]
      g0[3] = (unsigned)((ga >> 32) & 0x01FFFFFFu) | (2u << 30);  // global_addr[56:32] | type=2
      i32x8 g1;
      g1[0] = (int)((1u << 16) | (1u << 20) | (7u << 22) | (3u << 25)); // data_size=2B, pad_en, ivl=256dw, amt=4dw
      g1[1] = (int)(512u << 16);                             // tensor_dim0 = 512 (low 16)
      g1[2] = (int)(32u << 16);                              // tensor_dim0 hi=0 | tensor_dim1 = 32
      g1[3] = (int)(512u << 16);                             // tensor_dim1 hi=0 | tile_dim0 = 512
      g1[4] = 32;                                            // tile_dim1 = 32, tile_dim2 = 0
      g1[5] = 512;                                           // tensor_dim0_stride = 512 (low 32)
      g1[6] = 0;
      g1[7] = 0;
      i32x4 z4 = {0, 0, 0, 0};
#if defined(__clang_major__) && (__clang_major__ >= 23)
      i32x8 z8 = {0, 0, 0, 0, 0, 0, 0, 0};
      __builtin_amdgcn_tensor_load_to_lds(g0, g1, z4, z4, z8, 0);
#else
      __builtin_amdgcn_tensor_load_to_lds(g0, g1, z4, z4, 0);
#endif
    }

    // ---- stage x_t (f32 -> bf16, 32x64); overlaps with the TDM transfer ----
    for (int i = tid; i < (BATCH * INP) / 4; i += BDIM) {    // 512 x float4
      int b = i >> 4, c = i & 15;
      const float* xrow = x + ((size_t)b * T_STEPS + t) * INP;
      const float4 f = ((const float4*)xrow)[c];
      uint2 p; p.x = packbf(f.x, f.y); p.y = packbf(f.z, f.w);
      *(uint2*)(xb + b * 72 + c * 4) = p;
      if (t + 1 < T_STEPS) __builtin_prefetch(xrow + INP, 0, 0);  // next step's x
    }
    if (wv == 0) __builtin_amdgcn_s_wait_tensorcnt(0);
    __syncthreads();

    // ---- GEMM: this wave computes gates[0:32][wv*16 .. wv*16+16), K = 512(h) + 64(x) ----
    v8f acc0 = {0.f,0.f,0.f,0.f,0.f,0.f,0.f,0.f};   // batch rows 0..15
    v8f acc1 = {0.f,0.f,0.f,0.f,0.f,0.f,0.f,0.f};   // batch rows 16..31
    #pragma unroll 4
    for (int kc = 0; kc < 16; ++kc) {
      const uint4* bp = (const uint4*)(wbase + (size_t)kc * 256);
      uint4 q0 = bp[0], q1 = bp[1];
      v8u bu; bu[0]=q0.x; bu[1]=q0.y; bu[2]=q0.z; bu[3]=q0.w;
              bu[4]=q1.x; bu[5]=q1.y; bu[6]=q1.z; bu[7]=q1.w;
      // A fragment halves are 4 consecutive dwords each -> ds_load_b128
      uint4 a0lo = *(const uint4*)(ap0 + kc * 32);
      uint4 a0hi = *(const uint4*)(ap0 + kc * 32 + 16);
      uint4 a1lo = *(const uint4*)(ap1 + kc * 32);
      uint4 a1hi = *(const uint4*)(ap1 + kc * 32 + 16);
      v8u a0u; a0u[0]=a0lo.x; a0u[1]=a0lo.y; a0u[2]=a0lo.z; a0u[3]=a0lo.w;
               a0u[4]=a0hi.x; a0u[5]=a0hi.y; a0u[6]=a0hi.z; a0u[7]=a0hi.w;
      v8u a1u; a1u[0]=a1lo.x; a1u[1]=a1lo.y; a1u[2]=a1lo.z; a1u[3]=a1lo.w;
               a1u[4]=a1hi.x; a1u[5]=a1hi.y; a1u[6]=a1hi.z; a1u[7]=a1hi.w;
      acc0 = __builtin_amdgcn_wmma_f32_16x16x32_bf16(false, __builtin_bit_cast(v16bf, a0u),
                                                     false, __builtin_bit_cast(v16bf, bu),
                                                     (short)0, acc0, false, false);
      acc1 = __builtin_amdgcn_wmma_f32_16x16x32_bf16(false, __builtin_bit_cast(v16bf, a1u),
                                                     false, __builtin_bit_cast(v16bf, bu),
                                                     (short)0, acc1, false, false);
    }
    #pragma unroll
    for (int kc = 16; kc < 18; ++kc) {   // x contribution
      const uint4* bp = (const uint4*)(wbase + (size_t)kc * 256);
      uint4 q0 = bp[0], q1 = bp[1];
      v8u bu; bu[0]=q0.x; bu[1]=q0.y; bu[2]=q0.z; bu[3]=q0.w;
              bu[4]=q1.x; bu[5]=q1.y; bu[6]=q1.z; bu[7]=q1.w;
      int kx = (kc - 16) * 32;
      uint4 a0lo = *(const uint4*)(axp0 + kx);
      uint4 a0hi = *(const uint4*)(axp0 + kx + 16);
      uint4 a1lo = *(const uint4*)(axp1 + kx);
      uint4 a1hi = *(const uint4*)(axp1 + kx + 16);
      v8u a0u; a0u[0]=a0lo.x; a0u[1]=a0lo.y; a0u[2]=a0lo.z; a0u[3]=a0lo.w;
               a0u[4]=a0hi.x; a0u[5]=a0hi.y; a0u[6]=a0hi.z; a0u[7]=a0hi.w;
      v8u a1u; a1u[0]=a1lo.x; a1u[1]=a1lo.y; a1u[2]=a1lo.z; a1u[3]=a1lo.w;
               a1u[4]=a1hi.x; a1u[5]=a1hi.y; a1u[6]=a1hi.z; a1u[7]=a1hi.w;
      acc0 = __builtin_amdgcn_wmma_f32_16x16x32_bf16(false, __builtin_bit_cast(v16bf, a0u),
                                                     false, __builtin_bit_cast(v16bf, bu),
                                                     (short)0, acc0, false, false);
      acc1 = __builtin_amdgcn_wmma_f32_16x16x32_bf16(false, __builtin_bit_cast(v16bf, a1u),
                                                     false, __builtin_bit_cast(v16bf, bu),
                                                     (short)0, acc1, false, false);
    }

    // ---- scatter C tiles into gates LDS (C/D layout: vgpr r -> M = r + khalf*8) ----
    #pragma unroll
    for (int r = 0; r < 8; ++r) {
      gl[(mtop + r) * 132 + ncol]      = acc0[r];
      gl[(16 + mtop + r) * 132 + ncol] = acc1[r];
    }
    __syncthreads();

    // ---- LSTM nonlinearity + relaxed state update (each thread: 4 (b,hc) cells) ----
    {
      int e0 = tid * 4;
      int b  = e0 >> 5;
      int h0 = e0 & 31;
      #pragma unroll
      for (int j = 0; j < 4; ++j) {
        int hc = h0 + j;
        float ig = sigm (gl[b * 132 +       hc] + bl[      hc]);
        float fg = sigm (gl[b * 132 + 32  + hc] + bl[32  + hc]);
        float gg = tanhf(gl[b * 132 + 64  + hc] + bl[64  + hc]);
        float og = sigm (gl[b * 132 + 96  + hc] + bl[96  + hc]);
        float cp = sc[b * 32 + hc];
        float hp = sh[b * 32 + hc];
        float cn = fg * cp + ig * gg;
        float hn = og * tanhf(cn);
        float hr = hp + 0.5f * (hn - hp);
        float cr = cp + 0.5f * (cn - cp);
        sh[b * 32 + hc] = hr;
        sc[b * 32 + hc] = cr;
        out[((size_t)b * T_STEPS + t) * HID + wg * 32 + hc] = hr;
        wbuf[b * HID + wg * 32 + hc] = f2bf(hr);
        if (t == T_STEPS - 1) {
          out[finalOff + (size_t)b * 1024 +       wg * 32 + hc] = hr;
          out[finalOff + (size_t)b * 1024 + 512 + wg * 32 + hc] = cr;
        }
      }
    }

    grid_sync(cnt, (unsigned)((t + 1) * NWG));
  }
}

extern "C" void kernel_launch(void* const* d_in, const int* in_sizes, int n_in,
                              void* d_out, int out_size, void* d_ws, size_t ws_size,
                              hipStream_t stream) {
  const float* x    = (const float*)d_in[0];
  const float* W_ih = (const float*)d_in[1];
  const float* W_hh = (const float*)d_in[2];
  const float* b_ih = (const float*)d_in[3];
  const float* b_hh = (const float*)d_in[4];
  float* out = (float*)d_out;
  unsigned char* ws = (unsigned char*)d_ws;

  (void)in_sizes; (void)n_in; (void)out_size; (void)ws_size;

  hipFuncSetAttribute(reinterpret_cast<const void*>(stn_lstm_persistent),
                      hipFuncAttributeMaxDynamicSharedMemorySize, SMEM_BYTES);

  stn_init<<<32, 256, 0, stream>>>(ws);
  stn_lstm_persistent<<<NWG, BDIM, SMEM_BYTES, stream>>>(x, W_ih, W_hh, b_ih, b_hh, out, ws);
}
